// BoundaryLoss_58875411693877
// MI455X (gfx1250) — compile-verified, matrix operand order checked by reading, not run on hardware
//
#include <hip/hip_runtime.h>
#include <stdint.h>

#define B_IMGS   8
#define H        1024
#define W        1024
#define BH       16            // output rows per band
#define TROWS    (BH + 2)      // tile rows incl. vertical halo
#define BANDS    (H / BH)      // 64
#define NTHREADS 256           // 8 wave32

typedef __attribute__((ext_vector_type(16))) _Float16 v16h;
typedef __attribute__((ext_vector_type(8)))  float    v8f;

// --- CDNA5 async copy: global byte -> LDS, tracked on ASYNCcnt --------------
__device__ __forceinline__ void async_ld_b8(uint32_t lds_addr, const void* gaddr) {
    asm volatile("global_load_async_to_lds_b8 %0, %1, off"
                 :: "v"(lds_addr), "v"(gaddr)
                 : "memory");
}
__device__ __forceinline__ void wait_asynccnt0() {
    asm volatile("s_wait_asynccnt 0" ::: "memory");
}

__global__ __launch_bounds__(NTHREADS)
void boundary_band_kernel(const float* __restrict__ pred,
                          const int*   __restrict__ target,
                          int* __restrict__ colP,   // [B][W] pred-boundary column sums
                          int* __restrict__ colT,   // [B][W] target-boundary column sums
                          int* __restrict__ nPT)    // [B] joint counts
{
    __shared__ unsigned char smem[2 * TROWS * W];   // 36 KB
    unsigned char* tT = smem;                       // target class bytes
    unsigned char* tP = smem + TROWS * W;           // pred argmax bytes

    const int tid  = threadIdx.x;
    const int band = blockIdx.x;
    const int img  = blockIdx.y;
    const int r0   = band * BH;

    // ---- Stage target class bytes: async byte-gather straight into LDS ----
    const uint32_t ldsT0 = (uint32_t)(size_t)(void*)tT;
    for (int idx = tid; idx < TROWS * W; idx += NTHREADS) {
        int hh = idx >> 10;
        int w  = idx & (W - 1);
        int gh = r0 - 1 + hh;
        gh = gh < 0 ? 0 : (gh > H - 1 ? H - 1 : gh);      // edge replicate == -inf pad
        const char* gp = (const char*)(target + ((size_t)img * H + gh) * W + w);
        async_ld_b8(ldsT0 + (uint32_t)idx, gp);           // low byte of LE int32 = class
    }

    // ---- Stage pred argmax bytes (VALU), overlapping the async copies ----
    for (int idx = tid; idx < TROWS * W; idx += NTHREADS) {
        int hh = idx >> 10;
        int w  = idx & (W - 1);
        int gh = r0 - 1 + hh;
        gh = gh < 0 ? 0 : (gh > H - 1 ? H - 1 : gh);
        size_t base = ((size_t)img * 2 * H + gh) * W + w; // channel 0
        float p0 = pred[base];
        float p1 = pred[base + (size_t)H * W];            // channel 1
        tP[idx] = (p1 > p0) ? 1 : 0;                      // argmax (first max on tie)
    }

    wait_asynccnt0();
    __syncthreads();

    // ---- 3x3 boundary stencil: dilate=OR, erode=AND on packed (t | p<<1) ----
    int cntPT = 0;
    #pragma unroll
    for (int c = 0; c < 4; ++c) {                         // 4 columns per thread
        const int w  = (tid << 2) + c;
        const int wl = (w == 0)     ? 0     : w - 1;
        const int wr = (w == W - 1) ? W - 1 : w + 1;

        auto rowAgg = [&](int row, int& o, int& a) {
            int base = row * W;
            int v0 = (int)tT[base + wl] | ((int)tP[base + wl] << 1);
            int v1 = (int)tT[base + w ] | ((int)tP[base + w ] << 1);
            int v2 = (int)tT[base + wr] | ((int)tP[base + wr] << 1);
            o = v0 | v1 | v2;
            a = v0 & v1 & v2;
        };

        int o0, a0, o1, a1;
        rowAgg(0, o0, a0);
        rowAgg(1, o1, a1);
        int cT = 0, cP = 0;
        #pragma unroll
        for (int r = 0; r < BH; ++r) {
            int o2, a2;
            rowAgg(r + 2, o2, a2);
            int bnd = (o0 | o1 | o2) & ~(a0 & a1 & a2);   // bit0: target, bit1: pred
            cT += bnd & 1;
            int pb = (bnd >> 1) & 1;
            cP += pb;
            cntPT += pb & bnd;                            // pb & tb
            o0 = o1; a0 = a1; o1 = o2; a1 = a2;
        }
        atomicAdd(&colT[img * W + w], cT);
        atomicAdd(&colP[img * W + w], cP);
    }

    // ---- Joint-count wave reduction on the matrix pipe (ones-A WMMA) ----
    // D = ones(16x32) * B  => every D row holds B's column sums; sum of D[0]
    // over all 32 lanes = 2 * (sum of all B elements), layout-independent.
    v16h ones, bfrag;
    #pragma unroll
    for (int i = 0; i < 16; ++i) { ones[i] = (_Float16)1.0f; bfrag[i] = (_Float16)0.0f; }
    bfrag[0] = (_Float16)(float)cntPT;                    // <= 64, exact in f16
    v8f cz = {};
    v8f d = __builtin_amdgcn_wmma_f32_16x16x32_f16(
        /*neg_a=*/false, ones, /*neg_b=*/false, bfrag,
        /*c_mod=*/(short)0, cz, /*reuse_a=*/false, /*reuse_b=*/false);
    float s2 = d[0];
    #pragma unroll
    for (int off = 16; off > 0; off >>= 1)
        s2 += __shfl_xor(s2, off, 32);
    if ((tid & 31) == 0)
        atomicAdd(&nPT[img], (int)(s2 * 0.5f + 0.5f));
}

// ---- Closed-form BCE over {0,2} logits/labels + column-sum kill rule -------
__global__ __launch_bounds__(NTHREADS)
void finalize_kernel(const int* __restrict__ colP,
                     const int* __restrict__ colT,
                     const int* __restrict__ nPT,
                     float* __restrict__ out)
{
    __shared__ int sSumP, sKP, sKT;
    const int tid = threadIdx.x;
    double loss = 0.0;                                    // only tid 0 accumulates
    for (int b = 0; b < B_IMGS; ++b) {
        if (tid == 0) { sSumP = 0; sKP = 0; sKT = 0; }
        __syncthreads();
        int sp = 0, kp = 0, kt = 0;
        for (int w = tid; w < W; w += NTHREADS) {
            int cp = colP[b * W + w];
            int ct = colT[b * W + w];
            sp += cp;
            kp |= (cp >= 300);                            // LINE_THRESHOLD = 300
            kt |= (ct >= 300);
        }
        atomicAdd(&sSumP, sp);
        if (kp) atomicOr(&sKP, 1);
        if (kt) atomicOr(&sKT, 1);
        __syncthreads();
        if (tid == 0) {
            const double LN2 = 0.6931471805599453;        // term for x==0
            const double C2  = 2.1269280110429725;        // 2 + log1p(exp(-2))
            double P = sKP ? 0.0 : (double)sSumP;
            double J = (sKP || sKT) ? 0.0 : (double)nPT[b];
            loss += ((double)(H * (long long)W) - P) * LN2 + P * C2 - 4.0 * J;
        }
        __syncthreads();
    }
    if (tid == 0)
        out[0] = (float)(loss / (double)((long long)B_IMGS * H * W));
}

extern "C" void kernel_launch(void* const* d_in, const int* in_sizes, int n_in,
                              void* d_out, int out_size, void* d_ws, size_t ws_size,
                              hipStream_t stream) {
    const float* pred   = (const float*)d_in[0];
    const int*   target = (const int*)d_in[1];

    int* colP = (int*)d_ws;                 // 8*1024 ints
    int* colT = colP + B_IMGS * W;          // 8*1024 ints
    int* nPT  = colT + B_IMGS * W;          // 8 ints
    size_t zbytes = (size_t)(2 * B_IMGS * W + B_IMGS) * sizeof(int);
    hipMemsetAsync(d_ws, 0, zbytes, stream);

    dim3 grid(BANDS, B_IMGS);
    boundary_band_kernel<<<grid, NTHREADS, 0, stream>>>(pred, target, colP, colT, nPT);
    finalize_kernel<<<1, NTHREADS, 0, stream>>>(colP, colT, nPT, (float*)d_out);
}
